// DPCA3D_41016937676862
// MI455X (gfx1250) — matrix operand order repaired and checked by reading, not compiled, not run on hardware
//
#include <hip/hip_runtime.h>

typedef __attribute__((ext_vector_type(16))) _Float16 v16h;
typedef __attribute__((ext_vector_type(8)))  _Float16 v8h;
typedef __attribute__((ext_vector_type(8)))  float    v8f;
typedef __attribute__((ext_vector_type(4)))  int      v4i;

#define BATCH 2
#define C_DIM 128
#define D_DIM 16
#define H_DIM 32
#define W_DIM 32
#define NSPAT 16384            // D*H*W
#define NPOS  32768            // BATCH*NSPAT
#define HEADS 8
#define CH    64               // DIM_HEAD
#define INNER 512              // HEADS*CH
#define NKV   512              // 8*8*8 selected
#define BH    16               // BATCH*HEADS
#define NSPLIT 16              // q-row splits per (b,h) in attention

// ---------------------------------------------------------------------------
// async global->LDS 16B copy (CDNA5 GLOBAL_LOAD_ASYNC_TO_LDS_B128, ASYNCcnt)
// with a compile-safe fallback to a VGPR round-trip copy.
// Builtin signature (from hipcc diagnostic): param0 = AS1 v4i*, so pass
// explicitly address-space-qualified int4-vector pointers.
// ---------------------------------------------------------------------------
#if __has_builtin(__builtin_amdgcn_global_load_async_to_lds_b128)
#define HAVE_ASYNC_LDS 1
#else
#define HAVE_ASYNC_LDS 0
#endif

__device__ __forceinline__ void copy16_to_lds(const void* gsrc, void* lds_dst) {
#if HAVE_ASYNC_LDS
  __builtin_amdgcn_global_load_async_to_lds_b128(
      (__attribute__((address_space(1))) v4i*)gsrc,
      (__attribute__((address_space(3))) v4i*)lds_dst, 0, 0);
#else
  *(uint4*)lds_dst = *(const uint4*)gsrc;
#endif
}

__device__ __forceinline__ void wait_async_copies() {
#if HAVE_ASYNC_LDS
#if __has_builtin(__builtin_amdgcn_s_wait_asynccnt)
  __builtin_amdgcn_s_wait_asynccnt(0);
#else
  asm volatile("s_wait_asynccnt 0" ::: "memory");
#endif
#endif
}

// ---------------------------------------------------------------------------
// Kernel 1: channel layernorm (axis=1) of (b,c,D,H,W) fp32 -> (pos, c) f16
// ---------------------------------------------------------------------------
__global__ void ln_in_kernel(const float* __restrict__ x,
                             const float* __restrict__ g,
                             const float* __restrict__ be,
                             _Float16* __restrict__ y) {
  int p = blockIdx.x * blockDim.x + threadIdx.x;   // 0..NPOS
  if (p >= NPOS) return;
  int b = p >> 14, n = p & (NSPAT - 1);
  const float* base = x + (size_t)b * C_DIM * NSPAT + n;
  float s = 0.f, s2 = 0.f;
  #pragma unroll 4
  for (int c = 0; c < C_DIM; ++c) {
    float v = base[(size_t)c * NSPAT];
    s += v; s2 += v * v;
  }
  float mean = s * (1.f / C_DIM);
  float var  = s2 * (1.f / C_DIM) - mean * mean;
  float inv  = 1.f / (sqrtf(fmaxf(var, 0.f)) + 1e-6f);
  _Float16* dst = y + (size_t)p * C_DIM;
  #pragma unroll 4
  for (int c = 0; c < C_DIM; ++c) {
    float v = base[(size_t)c * NSPAT];
    dst[c] = (_Float16)(g[c] * (v - mean) * inv + be[c]);
  }
}

// ---------------------------------------------------------------------------
// Kernel 2: fp32 -> f16 convert (weights; (out,in) layout == Bt row-major)
// ---------------------------------------------------------------------------
__global__ void cvt_kernel(const float* __restrict__ src,
                           _Float16* __restrict__ dst, int n) {
  int i = blockIdx.x * blockDim.x + threadIdx.x;
  if (i < n) dst[i] = (_Float16)src[i];
}

// ---------------------------------------------------------------------------
// Kernel 3: WMMA GEMM  C(MxN) = A(MxK, f16 rowmajor) x Bt(NxK, f16 rowmajor)^T
// One wave -> 16x64 output tile.  f32 accumulate; out f16 or f32.
// ---------------------------------------------------------------------------
template <bool F32OUT>
__global__ __launch_bounds__(128) void gemm_wmma(const _Float16* __restrict__ A,
                                                 const _Float16* __restrict__ Bt,
                                                 void* __restrict__ Cout,
                                                 int M, int N, int K) {
  int wid  = blockIdx.x * (blockDim.x >> 5) + (threadIdx.x >> 5);
  int lane = threadIdx.x & 31;
  int n64  = N >> 6;
  int mt   = wid / n64, nt = wid % n64;
  if (mt >= (M >> 4)) return;
  int l15 = lane & 15, hl = lane >> 4;
  int row = (mt << 4) + l15;
  int colbase = nt << 6;
  v8f acc[4] = {};
  const _Float16* arow = A + (size_t)row * K;
  for (int k0 = 0; k0 < K; k0 += 32) {
    const _Float16* ap = arow + k0 + hl * 16;
    __builtin_prefetch(arow + k0 + 64, 0, 1);
    v8h alo = *(const v8h*)ap;
    v8h ahi = *(const v8h*)(ap + 8);
    v16h a;
    #pragma unroll
    for (int i = 0; i < 8; ++i) { a[i] = alo[i]; a[8 + i] = ahi[i]; }
    #pragma unroll
    for (int t = 0; t < 4; ++t) {
      int col = colbase + t * 16 + l15;
      v16h bv = *(const v16h*)(Bt + (size_t)col * K + k0 + hl * 16);
      acc[t] = __builtin_amdgcn_wmma_f32_16x16x32_f16(
          false, a, false, bv, (short)0, acc[t], false, false);
    }
  }
  #pragma unroll
  for (int t = 0; t < 4; ++t) {
    #pragma unroll
    for (int r = 0; r < 8; ++r) {
      int m = (mt << 4) + hl * 8 + r;
      int n = colbase + t * 16 + l15;
      if (F32OUT) ((float*)Cout)[(size_t)m * N + n] = acc[t][r];
      else        ((_Float16*)Cout)[(size_t)m * N + n] = (_Float16)acc[t][r];
    }
  }
}

// ---------------------------------------------------------------------------
// Kernel 4: per-head L2 normalize 64-ch segments in place
// ---------------------------------------------------------------------------
__global__ void l2n_kernel(_Float16* __restrict__ X, int nrows, int rowstride) {
  int it = blockIdx.x * blockDim.x + threadIdx.x;
  if (it >= nrows * HEADS) return;
  int row = it >> 3, hd = it & 7;
  _Float16* p = X + (size_t)row * rowstride + hd * CH;
  float s = 0.f;
  #pragma unroll 8
  for (int c = 0; c < CH; ++c) { float v = (float)p[c]; s += v * v; }
  float inv = 1.f / fmaxf(sqrtf(s), 1e-12f);
  #pragma unroll 8
  for (int c = 0; c < CH; ++c) p[c] = (_Float16)((float)p[c] * inv);
}

// ---------------------------------------------------------------------------
// Kernel 5: q_probe[bh][ch] = sum over spatial of normalized q
// ---------------------------------------------------------------------------
__global__ void qprobe_kernel(const _Float16* __restrict__ Q, float* __restrict__ qp) {
  int bh = blockIdx.x;
  int b = bh >> 3, h = bh & 7;
  int ch = threadIdx.x & 63, grp = threadIdx.x >> 6;   // 256 threads, 4 groups
  float s = 0.f;
  for (int n = grp; n < NSPAT; n += 4)
    s += (float)Q[((size_t)b * NSPAT + n) * INNER + h * CH + ch];
  __shared__ float red[256];
  red[threadIdx.x] = s;
  __syncthreads();
  if (grp == 0) qp[bh * CH + ch] = red[ch] + red[64 + ch] + red[128 + ch] + red[192 + ch];
}

// ---------------------------------------------------------------------------
// Kernel 6: axis |k| sums, scores, top-8 per axis (set selection suffices)
// ---------------------------------------------------------------------------
__global__ void topk_kernel(const _Float16* __restrict__ KV,
                            const float* __restrict__ qp,
                            int* __restrict__ idx) {
  int bh = blockIdx.x;
  int b = bh >> 3, h = bh & 7;
  const _Float16* kb = KV + (size_t)b * NSPAT * 1024 + h * CH;
  int ch = threadIdx.x;                       // 64 threads
  __shared__ float sd[64][16], shh[64][32], sww[64][32];
  for (int dd = 0; dd < 16; ++dd) {
    float s = 0.f;
    for (int i = 0; i < 1024; ++i) s += fabsf((float)kb[(size_t)(dd * 1024 + i) * 1024 + ch]);
    sd[ch][dd] = s;
  }
  for (int hh = 0; hh < 32; ++hh) {
    float s = 0.f;
    for (int d = 0; d < 16; ++d)
      for (int w = 0; w < 32; ++w) s += fabsf((float)kb[(size_t)(d * 1024 + hh * 32 + w) * 1024 + ch]);
    shh[ch][hh] = s;
  }
  for (int ww = 0; ww < 32; ++ww) {
    float s = 0.f;
    for (int d = 0; d < 16; ++d)
      for (int hh = 0; hh < 32; ++hh) s += fabsf((float)kb[(size_t)(d * 1024 + hh * 32 + ww) * 1024 + ch]);
    sww[ch][ww] = s;
  }
  __syncthreads();
  __shared__ float sc[80];    // [0..15]=d, [16..47]=h, [48..79]=w
  if (ch < 16) { float s = 0.f; for (int c = 0; c < 64; ++c) s += qp[bh * CH + c] * sd[c][ch];  sc[ch] = s; }
  if (ch < 32) {
    float s = 0.f, s2 = 0.f;
    for (int c = 0; c < 64; ++c) { s += qp[bh * CH + c] * shh[c][ch]; s2 += qp[bh * CH + c] * sww[c][ch]; }
    sc[16 + ch] = s; sc[48 + ch] = s2;
  }
  __syncthreads();
  if (ch < 3) {
    int cnt = (ch == 0) ? 16 : 32;
    int base = (ch == 0) ? 0 : ((ch == 1) ? 16 : 48);
    float tmp[32];
    for (int i = 0; i < cnt; ++i) tmp[i] = sc[base + i];
    for (int s8 = 0; s8 < 8; ++s8) {
      int bi = 0; float bv = tmp[0];
      for (int i = 1; i < cnt; ++i) if (tmp[i] > bv) { bv = tmp[i]; bi = i; }
      idx[bh * 24 + ch * 8 + s8] = bi;
      tmp[bi] = -1e30f;
    }
  }
}

// ---------------------------------------------------------------------------
// Kernel 7: gather selected k rows -> kf (bh,512,64) and v -> vT (bh,64,512)
// ---------------------------------------------------------------------------
__global__ void gather_kernel(const _Float16* __restrict__ KV,
                              const int* __restrict__ idx,
                              _Float16* __restrict__ kf_g,
                              _Float16* __restrict__ vt_g) {
  int bh = blockIdx.x;
  int b = bh >> 3, h = bh & 7;
  for (int j = threadIdx.x; j < NKV; j += blockDim.x) {
    int a = j >> 6, bb = (j >> 3) & 7, cc = j & 7;
    int n = idx[bh * 24 + a] * 1024 + idx[bh * 24 + 8 + bb] * 32 + idx[bh * 24 + 16 + cc];
    const _Float16* src = KV + ((size_t)b * NSPAT + n) * 1024 + h * CH;
    _Float16* kd = kf_g + ((size_t)bh * NKV + j) * CH;
    #pragma unroll 8
    for (int c = 0; c < CH; ++c) {
      kd[c] = src[c];
      vt_g[((size_t)bh * CH + c) * NKV + j] = src[INNER + c];
    }
  }
}

// ---------------------------------------------------------------------------
// Kernel 8: cosine attention.  sim = q.k in [-1,1] -> softmax via exp(s-1).
// kf/vT brought to LDS with async global->LDS (ASYNCcnt); WMMA for both
// GEMMs; per-wave LDS staging for the P (C-layout -> A-layout) transpose.
// ---------------------------------------------------------------------------
__global__ __launch_bounds__(256) void attn_kernel(const _Float16* __restrict__ Q,
                                                   const _Float16* __restrict__ kf_g,
                                                   const _Float16* __restrict__ vt_g,
                                                   _Float16* __restrict__ AO) {
  int bh = blockIdx.x, split = blockIdx.y;
  int b = bh >> 3, h = bh & 7;
  __shared__ __align__(32) _Float16 kf_s[NKV * CH];     // 64 KB
  __shared__ __align__(32) _Float16 vt_s[CH * NKV];     // 64 KB
  __shared__ __align__(16) _Float16 pbuf[8][16 * 40];   // 10 KB, per-wave
  {
    const uint4* s1 = (const uint4*)(kf_g + (size_t)bh * NKV * CH);
    uint4* d1 = (uint4*)kf_s;
    const uint4* s2 = (const uint4*)(vt_g + (size_t)bh * CH * NKV);
    uint4* d2 = (uint4*)vt_s;
    for (int i = threadIdx.x; i < NKV * CH * 2 / 16; i += 256) {
      copy16_to_lds(s1 + i, d1 + i);
      copy16_to_lds(s2 + i, d2 + i);
    }
    wait_async_copies();
  }
  __syncthreads();

  int wv = threadIdx.x >> 5, lane = threadIdx.x & 31;
  int hl = lane >> 4, l15 = lane & 15;
  const int rows_per_block = NSPAT / NSPLIT;            // 1024
  const int tiles = rows_per_block / 16;                // 64

  for (int tile = wv; tile < tiles; tile += 8) {
    int n0 = split * rows_per_block + tile * 16;
    const _Float16* qrow = Q + ((size_t)b * NSPAT + n0 + l15) * INNER + h * CH;
    v16h aq[2];
    #pragma unroll
    for (int kk = 0; kk < 2; ++kk) {
      v8h lo = *(const v8h*)(qrow + kk * 32 + hl * 16);
      v8h hi = *(const v8h*)(qrow + kk * 32 + 8 + hl * 16);
      #pragma unroll
      for (int i = 0; i < 8; ++i) { aq[kk][i] = lo[i]; aq[kk][8 + i] = hi[i]; }
    }
    v8f acc[4] = {};
    float rs[8] = {};
    for (int j0 = 0; j0 < NKV; j0 += 32) {
      #pragma unroll
      for (int st = 0; st < 2; ++st) {
        int col = j0 + st * 16 + l15;
        v8f s = {};
        #pragma unroll
        for (int kk = 0; kk < 2; ++kk) {
          v16h bk = *(const v16h*)(kf_s + (size_t)col * CH + kk * 32 + hl * 16);
          s = __builtin_amdgcn_wmma_f32_16x16x32_f16(
              false, aq[kk], false, bk, (short)0, s, false, false);
        }
        #pragma unroll
        for (int r = 0; r < 8; ++r) {
          float p = __expf(s[r] - 1.0f);     // sim <= 1 (cosine), overflow-free
          rs[r] += p;
          pbuf[wv][(hl * 8 + r) * 40 + st * 16 + l15] = (_Float16)p;
        }
      }
      // reload P in A-matrix fragment layout
      const _Float16* pp = &pbuf[wv][l15 * 40 + hl * 16];
      v8h plo = *(const v8h*)pp;
      v8h phi = *(const v8h*)(pp + 8);
      v16h pa;
      #pragma unroll
      for (int i = 0; i < 8; ++i) { pa[i] = plo[i]; pa[8 + i] = phi[i]; }
      #pragma unroll
      for (int ct = 0; ct < 4; ++ct) {
        v16h bv = *(const v16h*)(vt_s + (size_t)(ct * 16 + l15) * NKV + j0 + hl * 16);
        acc[ct] = __builtin_amdgcn_wmma_f32_16x16x32_f16(
            false, pa, false, bv, (short)0, acc[ct], false, false);
      }
    }
    float rinv[8];
    #pragma unroll
    for (int r = 0; r < 8; ++r) {
      float v = rs[r];
      v += __shfl_xor(v, 1, 32);
      v += __shfl_xor(v, 2, 32);
      v += __shfl_xor(v, 4, 32);
      v += __shfl_xor(v, 8, 32);
      rinv[r] = 1.f / v;
    }
    #pragma unroll
    for (int ct = 0; ct < 4; ++ct)
      #pragma unroll
      for (int r = 0; r < 8; ++r)
        AO[((size_t)b * NSPAT + n0 + hl * 8 + r) * INNER + h * CH + ct * 16 + l15] =
            (_Float16)(acc[ct][r] * rinv[r]);
  }
}

// ---------------------------------------------------------------------------
// Kernel 9: out LN (over 128 ch) + gamma*out + residual, to (b,c,D,H,W) fp32
// ---------------------------------------------------------------------------
__global__ void ln_out_kernel(const float* __restrict__ O32,
                              const float* __restrict__ qs,
                              const float* __restrict__ g,
                              const float* __restrict__ be,
                              const float* __restrict__ gam,
                              float* __restrict__ out) {
  int row = blockIdx.x;          // 0..NPOS
  int c = threadIdx.x;           // 128
  float v = O32[(size_t)row * C_DIM + c];
  __shared__ float r1[128], r2[128];
  r1[c] = v; r2[c] = v * v;
  __syncthreads();
  for (int s = 64; s > 0; s >>= 1) {
    if (c < s) { r1[c] += r1[c + s]; r2[c] += r2[c + s]; }
    __syncthreads();
  }
  float mean = r1[0] * (1.f / C_DIM);
  float var  = r2[0] * (1.f / C_DIM) - mean * mean;
  float inv  = 1.f / (sqrtf(fmaxf(var, 0.f)) + 1e-6f);
  float y = g[c] * (v - mean) * inv + be[c];
  int b = row >> 14, n = row & (NSPAT - 1);
  size_t o = ((size_t)b * C_DIM + c) * NSPAT + n;
  out[o] = gam[0] * y + qs[o];
}

// ---------------------------------------------------------------------------
extern "C" void kernel_launch(void* const* d_in, const int* in_sizes, int n_in,
                              void* d_out, int out_size, void* d_ws, size_t ws_size,
                              hipStream_t stream) {
  const float* qs    = (const float*)d_in[0];
  const float* ctx   = (const float*)d_in[1];
  const float* w_q   = (const float*)d_in[2];
  const float* w_kv  = (const float*)d_in[3];
  const float* w_out = (const float*)d_in[4];
  const float* ctx_g = (const float*)d_in[5];
  const float* ctx_b = (const float*)d_in[6];
  const float* qs_g  = (const float*)d_in[7];
  const float* qs_b  = (const float*)d_in[8];
  const float* out_g = (const float*)d_in[9];
  const float* out_b = (const float*)d_in[10];
  const float* gam   = (const float*)d_in[11];
  float* out = (float*)d_out;

  char* ws = (char*)d_ws;
  // workspace overlays (total ~114.5 MB):
  //  [0,16MB)    ctxn(8MB)+qsn(8MB)  -> later O32 (16MB)
  //  [16,48MB)   Q f16 (32MB)
  //  [48,112MB)  KV f16 (64MB)       -> later AO (32MB)
  //  [112MB,..)  weights + probe + idx + kf + vT
  _Float16* ctxn = (_Float16*)(ws);
  _Float16* qsn  = (_Float16*)(ws + ((size_t)8 << 20));
  _Float16* Qb   = (_Float16*)(ws + ((size_t)16 << 20));
  _Float16* KVb  = (_Float16*)(ws + ((size_t)48 << 20));
  _Float16* AO   = (_Float16*)(ws + ((size_t)48 << 20));   // aliases KV (dead)
  float*    O32  = (float*)(ws);                           // aliases ctxn/qsn (dead)
  size_t Wo = (size_t)112 << 20;
  _Float16* wq16  = (_Float16*)(ws + Wo);
  _Float16* wkv16 = (_Float16*)(ws + Wo + 0x20000);
  _Float16* wo16  = (_Float16*)(ws + Wo + 0x60000);
  float*    qp    = (float*)(ws + Wo + 0x80000);
  int*      idx   = (int*)(ws + Wo + 0x81000);
  _Float16* kf_g  = (_Float16*)(ws + Wo + 0x82000);
  _Float16* vt_g  = (_Float16*)(ws + Wo + 0x182000);

  // 1) channel layernorms
  ln_in_kernel<<<NPOS / 256, 256, 0, stream>>>(ctx, ctx_g, ctx_b, ctxn);
  ln_in_kernel<<<NPOS / 256, 256, 0, stream>>>(qs, qs_g, qs_b, qsn);

  // 2) weight f32->f16 (layouts already Bt row-major)
  cvt_kernel<<<(INNER * C_DIM + 255) / 256, 256, 0, stream>>>(w_q, wq16, INNER * C_DIM);
  cvt_kernel<<<(2 * INNER * C_DIM + 255) / 256, 256, 0, stream>>>(w_kv, wkv16, 2 * INNER * C_DIM);
  cvt_kernel<<<(C_DIM * INNER + 255) / 256, 256, 0, stream>>>(w_out, wo16, C_DIM * INNER);

  // 3) projections (WMMA):  Q = qsn x Wq^T ;  KV = ctxn x Wkv^T
  gemm_wmma<false><<<(NPOS / 16) * (INNER / 64) / 4, 128, 0, stream>>>(qsn, wq16, Qb, NPOS, INNER, C_DIM);
  gemm_wmma<false><<<(NPOS / 16) * (2 * INNER / 64) / 4, 128, 0, stream>>>(ctxn, wkv16, KVb, NPOS, 2 * INNER, C_DIM);

  // 4) l2norm q and k (per-head 64-ch segments)
  l2n_kernel<<<(NPOS * HEADS + 255) / 256, 256, 0, stream>>>(Qb, NPOS, INNER);
  l2n_kernel<<<(NPOS * HEADS + 255) / 256, 256, 0, stream>>>(KVb, NPOS, 2 * INNER);

  // 5) q_probe, top-k per axis, gather selected k/v
  qprobe_kernel<<<BH, 256, 0, stream>>>(Qb, qp);
  topk_kernel<<<BH, 64, 0, stream>>>(KVb, qp, idx);
  gather_kernel<<<BH, 256, 0, stream>>>(KVb, idx, kf_g, vt_g);

  // 6) attention (LDS-resident k/v via async copy, WMMA, max-free softmax)
  attn_kernel<<<dim3(BH, NSPLIT), 256, 0, stream>>>(Qb, kf_g, vt_g, AO);

  // 7) output projection (WMMA, f32 out)
  gemm_wmma<true><<<(NPOS / 16) * (C_DIM / 64) / 4, 128, 0, stream>>>(AO, wo16, O32, NPOS, C_DIM, INNER);

  // 8) final LN + gamma*out + residual
  ln_out_kernel<<<NPOS, 128, 0, stream>>>(O32, qs, out_g, out_b, gam, out);
}
